// CAL_GCN_66752381714633
// MI455X (gfx1250) — compile-verified, matrix operand order checked by reading, not run on hardware
//
#include <hip/hip_runtime.h>
#include <hip/hip_bf16.h>

// ---------------------------------------------------------------------------
// CAL-GCN forward for MI455X (gfx1250, wave32, WMMA).
// GEMMs use v_wmma_f32_16x16x32_bf16 with BN folded into the A-tile load as a
// per-column affine (scale/shift). Weights are pre-packed into WMMA fragment
// order so each B fragment is one 32-byte v16bf load (global_load_b128 pairs);
// all four B fragments are preloaded per k-step to avoid WMMA WAR stalls.
// Edge-rep concat is gathered inside the GEMM. Scatter segment-sums use
// float4-vectorized f32 atomics (bandwidth bound). Workspace ~920 MB.
// ---------------------------------------------------------------------------

typedef __attribute__((ext_vector_type(16))) __bf16 v16bf;
typedef __attribute__((ext_vector_type(8)))  float  v8f;

static constexpr int  H_   = 128;
static constexpr long N_   = 200000;
static constexpr long C_   = 50000;
static constexpr long NC_  = N_ + C_;
static constexpr long B_   = 4096;
static constexpr long E_   = 800000;
static constexpr long EHI_ = 1200000;
static constexpr long M_   = 400000;

// ---- workspace layout (bytes) ----
static constexpr unsigned long long SZ_NC  = (unsigned long long)NC_ * H_ * 4;   // 128 MB
static constexpr unsigned long long SZ_N   = (unsigned long long)N_  * H_ * 4;   // 102.4 MB
static constexpr unsigned long long OFF_X   = 0;
static constexpr unsigned long long OFF_B1  = OFF_X  + SZ_NC;   // msg (hi) / xs / hacc2(xs)
static constexpr unsigned long long OFF_B2  = OFF_B1 + SZ_NC;   // hacc (hi) / xc / hacc2(xc)
static constexpr unsigned long long OFF_B3  = OFF_B2 + SZ_NC;   // hidC (motif) / msg2 (stage2)
static constexpr unsigned long long OFF_HID = OFF_B3 + SZ_N;    // E x 256 bf16 edge hidden
static constexpr unsigned long long SZ_HID  = (unsigned long long)E_ * 256 * 2;  // 409.6 MB
static constexpr unsigned long long OFF_SM  = OFF_HID + SZ_HID;
// small region (cursor offsets from OFF_SM)
static constexpr unsigned long long SM_DEGHI  = 0;                         // NC f32
static constexpr unsigned long long SM_DINVHI = SM_DEGHI  + NC_ * 4;
static constexpr unsigned long long SM_CSUM   = SM_DINVHI + NC_ * 4;       // 256 f32
static constexpr unsigned long long SM_CSQ    = SM_CSUM   + 1024;
static constexpr unsigned long long SM_AFF    = SM_CSQ    + 1024;          // 512 f32
static constexpr unsigned long long SM_WBF    = SM_AFF    + 2048;          // 256*256 bf16 (packed)
static constexpr unsigned long long SM_RCNT   = SM_WBF    + 131072;        // N f32
static constexpr unsigned long long SM_CCNT   = SM_RCNT   + N_ * 4;
static constexpr unsigned long long SM_DEGS   = SM_CCNT   + N_ * 4;
static constexpr unsigned long long SM_DINVS  = SM_DEGS   + N_ * 4;
static constexpr unsigned long long SM_NM     = SM_DINVS  + N_ * 4;        // N*2 f32
static constexpr unsigned long long SM_EA     = SM_NM     + N_ * 8;        // E*2 f32
static constexpr unsigned long long SM_GMEAN  = SM_EA     + E_ * 8;        // B*128 f32
static constexpr unsigned long long SM_CNT    = SM_GMEAN  + B_ * H_ * 4;
static constexpr unsigned long long SM_EPSB   = SM_CNT    + B_ * 4;
static constexpr unsigned long long SM_SXS    = SM_EPSB   + B_ * 4;        // B*128
static constexpr unsigned long long SM_SXC    = SM_SXS    + B_ * H_ * 4;
static constexpr unsigned long long SM_SSUM   = SM_SXC    + B_ * H_ * 4;
static constexpr unsigned long long SM_HIDB   = SM_SSUM   + B_ * H_ * 4;
static constexpr unsigned long long SM_NA     = SM_HIDB   + B_ * H_ * 4;   // C*2 f32

// ---------------------------------------------------------------------------
// generic helpers
// ---------------------------------------------------------------------------
__global__ void k_fill(float* p, long n, float v) {
  long t = blockIdx.x * (long)blockDim.x + threadIdx.x;
  for (; t < n; t += (long)gridDim.x * blockDim.x) p[t] = v;
}

__global__ void k_count(const int* idx, long n, float* cnt) {
  long t = blockIdx.x * (long)blockDim.x + threadIdx.x;
  for (; t < n; t += (long)gridDim.x * blockDim.x) atomicAdd(&cnt[idx[t]], 1.0f);
}

__global__ void k_deg_ew(const int* dst, const float* ew, int stride, long n, float* deg) {
  long t = blockIdx.x * (long)blockDim.x + threadIdx.x;
  for (; t < n; t += (long)gridDim.x * blockDim.x)
    atomicAdd(&deg[dst[t]], ew[t * stride]);
}

__global__ void k_rsqrt(const float* deg, float* dinv, long n) {
  long t = blockIdx.x * (long)blockDim.x + threadIdx.x;
  for (; t < n; t += (long)gridDim.x * blockDim.x)
    dinv[t] = rsqrtf(fmaxf(deg[t], 1e-12f));
}

__global__ void k_addvec(const float* a, const float* b, float* o, long n) {
  long t = blockIdx.x * (long)blockDim.x + threadIdx.x;
  for (; t < n; t += (long)gridDim.x * blockDim.x) o[t] = a[t] + b[t];
}

// ---------------------------------------------------------------------------
// Pack W (KxNout f32, row-major) into WMMA B-fragment order:
//   Wp[(((kt*Nout + col)*2 + half)*16) + v] = bf16( W[k, col] )
// where kt = k-tile (32 K per tile), half selects lane group, and element v
// maps to k = kt*32 + half*8 + (v<8 ? v : v+8)  (ISA 16-bit B layout).
// Each GEMM lane then loads its whole fragment as one contiguous 32B v16bf.
// ---------------------------------------------------------------------------
__global__ void k_pack_w(const float* __restrict__ W, __bf16* __restrict__ Wp,
                         int K, int Nout) {
  long t = blockIdx.x * (long)blockDim.x + threadIdx.x;
  const long total = (long)K * Nout;
  for (; t < total; t += (long)gridDim.x * blockDim.x) {
    const int  v    = (int)(t & 15);
    const int  half = (int)((t >> 4) & 1);
    const long rest = t >> 5;
    const int  col  = (int)(rest % Nout);
    const long kt   = rest / Nout;
    const long k    = kt * 32 + half * 8 + (v < 8 ? v : v + 8);
    Wp[t] = (__bf16)W[k * Nout + col];
  }
}

// ---------------------------------------------------------------------------
// BN column statistics (two-pass, block-partial + f32 atomics)
// blockDim.x == cols (128 or 256); coalesced row-wise reads.
// ---------------------------------------------------------------------------
__global__ void k_colstats_f32(const float* X, long rows, float* sum, float* sq) {
  const int c = threadIdx.x, cols = blockDim.x;
  float s = 0.f, q = 0.f;
  for (long r = blockIdx.x; r < rows; r += gridDim.x) {
    float v = X[r * (long)cols + c];
    s += v; q += v * v;
  }
  atomicAdd(&sum[c], s); atomicAdd(&sq[c], q);
}

__global__ void k_colstats_bf16(const __bf16* X, long rows, float* sum, float* sq) {
  const int c = threadIdx.x, cols = blockDim.x;
  float s = 0.f, q = 0.f;
  for (long r = blockIdx.x; r < rows; r += gridDim.x) {
    float v = (float)X[r * (long)cols + c];
    s += v; q += v * v;
  }
  atomicAdd(&sum[c], s); atomicAdd(&sq[c], q);
}

// edge_rep column stats in O(N*H): weight node columns by edge-endpoint counts
__global__ void k_wcolstats(const float* xn, const float* rcnt, const float* ccnt,
                            float* sum, float* sq) {
  const int t = threadIdx.x;            // 0..255
  const int c = t & 127;
  const float* w = (t < 128) ? rcnt : ccnt;
  float s = 0.f, q = 0.f;
  for (long v = blockIdx.x; v < N_; v += gridDim.x) {
    float wv = w[v], xv = xn[v * 128 + c];
    s += wv * xv; q += wv * xv * xv;
  }
  atomicAdd(&sum[t], s); atomicAdd(&sq[t], q);
}

// fold BN into per-column affine: bn(x) = aff[c]*x + aff[cols+c]
__global__ void k_aff(const float* sum, const float* sq, const float* g, const float* b,
                      float inv_rows, int cols, float* aff) {
  int c = threadIdx.x;
  if (c < cols) {
    float m  = sum[c] * inv_rows;
    float vr = sq[c] * inv_rows - m * m;
    float sc = g[c] * rsqrtf(vr + 1e-5f);
    aff[c] = sc;
    aff[cols + c] = b[c] - m * sc;
  }
}

// ---------------------------------------------------------------------------
// WMMA GEMM:  Y[r, j] = relu?( sum_k (aff_k * X[gather(r), k] + shift_k) * W[k, j] + bias_j )
// A is bf16-converted on the fly from f32 (float4 vector loads; 8-element
// K-groups never straddle the 128-column concat boundary). The four B
// fragments are preloaded into separate registers via incremental pointers
// (one clause of 8x global_load_b128), then the 4 WMMAs issue back-to-back.
// rows%16==0, K%32==0, Nout%64==0 -> no predication (EXEC all-ones for WMMA).
// ---------------------------------------------------------------------------
__global__ __launch_bounds__(32) void k_gemm_aff_wmma(
    const float* __restrict__ X, const int* __restrict__ idx0, const int* __restrict__ idx1,
    int K, int Nout, const float* __restrict__ aff,
    const __bf16* __restrict__ Wp, const float* __restrict__ bias,
    float* __restrict__ Yf, __bf16* __restrict__ Ybf, int relu)
{
  const int lane = threadIdx.x;
  const int n    = lane & 15;        // A row within tile / B,C column within tile
  const int half = lane >> 4;        // K-half select (ISA 16-bit A/B layout)
  const long rowA    = (long)blockIdx.x * 16 + n;
  const int  colBase = blockIdx.y * 64;

  const float* rs0; const float* rs1;
  if (idx0) {
    rs0 = X + (long)idx0[rowA] * 128;
    rs1 = X + (long)idx1[rowA] * 128 - 128;   // pre-biased: rs1[k] == X[idx1][k-128]
  } else {
    rs0 = X + rowA * (long)K;
    rs1 = rs0;
  }

  // incremental B-fragment pointers (32B fragment per (col,half) per k-tile)
  const __bf16* bp[4];
  #pragma unroll
  for (int t = 0; t < 4; ++t) {
    const long col = colBase + t * 16 + n;
    bp[t] = Wp + ((col * 2 + half) << 4);
  }
  const long bstep = (long)Nout * 32;   // bf16 elements per k-tile

  v8f acc[4];
  #pragma unroll
  for (int t = 0; t < 4; ++t)
    #pragma unroll
    for (int r = 0; r < 8; ++r) acc[t][r] = 0.0f;

  for (int kb = 0; kb < K; kb += 32) {
    // start all four B-fragment loads first (one clause, overlaps A math)
    v16bf bq[4];
    #pragma unroll
    for (int t = 0; t < 4; ++t) {
      bq[t] = *(const v16bf*)bp[t];
      bp[t] += bstep;
    }
    __builtin_prefetch(bp[0], 0, 0);   // global_prefetch_b8: next k-tile of W
    // A fragment: element v<8 -> k = kb + half*8 + v ; v>=8 -> k = kb+16+half*8+(v-8)
    v16bf a;
    #pragma unroll
    for (int g = 0; g < 2; ++g) {
      const int kb8 = kb + g * 16 + half * 8;        // 8-aligned K-group base
      const float* px = ((kb8 < 128) ? rs0 : rs1) + kb8;
      const float4 xv0 = *(const float4*)(px);
      const float4 xv1 = *(const float4*)(px + 4);
      const float4 sc0 = *(const float4*)(aff + kb8);
      const float4 sc1 = *(const float4*)(aff + kb8 + 4);
      const float4 sh0 = *(const float4*)(aff + K + kb8);
      const float4 sh1 = *(const float4*)(aff + K + kb8 + 4);
      const int o = g * 8;
      a[o + 0] = (__bf16)(sc0.x * xv0.x + sh0.x);
      a[o + 1] = (__bf16)(sc0.y * xv0.y + sh0.y);
      a[o + 2] = (__bf16)(sc0.z * xv0.z + sh0.z);
      a[o + 3] = (__bf16)(sc0.w * xv0.w + sh0.w);
      a[o + 4] = (__bf16)(sc1.x * xv1.x + sh1.x);
      a[o + 5] = (__bf16)(sc1.y * xv1.y + sh1.y);
      a[o + 6] = (__bf16)(sc1.z * xv1.z + sh1.z);
      a[o + 7] = (__bf16)(sc1.w * xv1.w + sh1.w);
    }
    // 4 back-to-back WMMAs, no WAR on B sources (separate regs per fragment)
    #pragma unroll
    for (int t = 0; t < 4; ++t)
      acc[t] = __builtin_amdgcn_wmma_f32_16x16x32_bf16(
          false, a, false, bq[t], (short)0, acc[t], false, false);
  }

  // C/D layout: lanes 0-15 N=lane M=r; lanes 16-31 N=lane-16 M=r+8
  #pragma unroll
  for (int t = 0; t < 4; ++t) {
    const int col = colBase + t * 16 + n;
    const float bv = bias ? bias[col] : 0.0f;
    #pragma unroll
    for (int r = 0; r < 8; ++r) {
      const long row = (long)blockIdx.x * 16 + half * 8 + r;
      float v = acc[t][r] + bv;
      if (relu) v = fmaxf(v, 0.0f);
      const long o = row * (long)Nout + col;
      if (Yf)  Yf[o]  = v;
      if (Ybf) Ybf[o] = (__bf16)v;
    }
  }
}

// ---------------------------------------------------------------------------
// graph kernels
// ---------------------------------------------------------------------------
struct EmbP { const float* p[9]; };

__global__ void k_embed(EmbP emb, const int* __restrict__ xa, float* __restrict__ x0) {
  const long n = blockIdx.x; const int h = threadIdx.x;
  __shared__ int idx[9];
  if (h < 9) idx[h] = xa[n * 9 + h];
  __syncthreads();
  float s = 0.f;
  #pragma unroll
  for (int i = 0; i < 9; ++i) s += emb.p[i][(long)idx[i] * 128 + h];
  x0[n * 128 + h] = s;
}

__global__ void k_pool(const int* __restrict__ cmn, const int* __restrict__ cmc,
                       const float* __restrict__ x0, float* __restrict__ xpool) {
  long t = blockIdx.x * (long)blockDim.x + threadIdx.x;
  const long total = M_ * 128;
  for (; t < total; t += (long)gridDim.x * blockDim.x) {
    const long m = t >> 7; const int h = (int)(t & 127);
    atomicAdd(&xpool[(long)cmc[m] * 128 + h], x0[(long)cmn[m] * 128 + h]);
  }
}

// hacc[dst] += dinv[src]*w*dinv[dst] * msg[src]; float4-vectorized (4 dims/thread)
__global__ void k_scatter(const int* __restrict__ src, const int* __restrict__ dst,
                          const float* __restrict__ ew, int ewstride,
                          const float* __restrict__ dinv, const float* __restrict__ msg,
                          float* __restrict__ hacc, long nE) {
  long t = blockIdx.x * (long)blockDim.x + threadIdx.x;
  const long total = nE * 32;
  for (; t < total; t += (long)gridDim.x * blockDim.x) {
    const long e = t >> 5; const int h = (int)(t & 31) * 4;
    const int s = src[e], d = dst[e];
    const float w = ew ? ew[e * ewstride] : 1.0f;
    const float coef = dinv[s] * w * dinv[d];
    const float4 mv = *(const float4*)(msg + (long)s * 128 + h);
    float* hp = hacc + (long)d * 128 + h;
    atomicAdd(hp + 0, coef * mv.x);
    atomicAdd(hp + 1, coef * mv.y);
    atomicAdd(hp + 2, coef * mv.z);
    atomicAdd(hp + 3, coef * mv.w);
  }
}

// residual GCN epilogue: x += relu(hacc + dinv^2 * msg + bias)
__global__ void k_hi_fin(float* __restrict__ x, const float* __restrict__ hacc,
                         const float* __restrict__ msg, const float* __restrict__ dinv,
                         const float* __restrict__ bias) {
  long t = blockIdx.x * (long)blockDim.x + threadIdx.x;
  const long total = NC_ * 128;
  for (; t < total; t += (long)gridDim.x * blockDim.x) {
    const long v = t >> 7; const int h = (int)(t & 127);
    const float di = dinv[v];
    x[t] += fmaxf(hacc[t] + di * di * msg[t] + bias[h], 0.0f);
  }
}

// stage-2 epilogue fused with batch segment-sum
__global__ void k_s2_fin(const float* __restrict__ hacc, const float* __restrict__ msg,
                         const float* __restrict__ dinv, const float* __restrict__ bias,
                         const int* __restrict__ batch, float* __restrict__ S) {
  long t = blockIdx.x * (long)blockDim.x + threadIdx.x;
  const long total = N_ * 128;
  for (; t < total; t += (long)gridDim.x * blockDim.x) {
    const long v = t >> 7; const int h = (int)(t & 127);
    const float di = dinv[v];
    const float val = fmaxf(hacc[t] + di * di * msg[t] + bias[h], 0.0f);
    atomicAdd(&S[(long)batch[v] * 128 + h], val);
  }
}

// MLP head (K -> 2), raw logits
__global__ void k_head(const float* __restrict__ X, const float* __restrict__ aff,
                       const float* __restrict__ W2, const float* __restrict__ c2,
                       int K, long rows, float* __restrict__ out) {
  long t = blockIdx.x * (long)blockDim.x + threadIdx.x;
  const long total = rows * 2;
  for (; t < total; t += (long)gridDim.x * blockDim.x) {
    const long r = t >> 1; const int o = (int)(t & 1);
    float acc = c2[o];
    for (int c = 0; c < K; ++c)
      acc += (aff[c] * X[r * (long)K + c] + aff[K + c]) * W2[c * 2 + o];
    out[r * 2 + o] = acc;
  }
}

// edge head: bn2(hid) @ W2 + c2, then 2-way softmax
__global__ void k_edge_head(const __bf16* __restrict__ hid, const float* __restrict__ aff,
                            const float* __restrict__ W2, const float* __restrict__ c2,
                            float* __restrict__ ea) {
  long e = blockIdx.x * (long)blockDim.x + threadIdx.x;
  for (; e < E_; e += (long)gridDim.x * blockDim.x) {
    float l0 = c2[0], l1 = c2[1];
    for (int c = 0; c < 256; ++c) {
      const float hv = aff[c] * (float)hid[e * 256 + c] + aff[256 + c];
      l0 += hv * W2[c * 2 + 0];
      l1 += hv * W2[c * 2 + 1];
    }
    const float m = fmaxf(l0, l1);
    const float e0 = expf(l0 - m), e1 = expf(l1 - m);
    const float inv = 1.0f / (e0 + e1);
    ea[e * 2 + 0] = e0 * inv;
    ea[e * 2 + 1] = e1 * inv;
  }
}

// nm = softmax(node_att + gumbel1)[.,0]; nm_ = softmax(node_att + gumbel2)[.,1]
__global__ void k_nm(const float* __restrict__ na, const int* __restrict__ c2n,
                     const float* __restrict__ g1, const float* __restrict__ g2,
                     float* __restrict__ nm) {
  long n = blockIdx.x * (long)blockDim.x + threadIdx.x;
  for (; n < N_; n += (long)gridDim.x * blockDim.x) {
    const long cq = c2n[n];
    const float a0 = na[cq * 2], a1 = na[cq * 2 + 1];
    const float x0 = a0 + g1[n * 2], x1 = a1 + g1[n * 2 + 1];
    const float y0 = a0 + g2[n * 2], y1 = a1 + g2[n * 2 + 1];
    nm[n * 2 + 0] = 1.0f / (1.0f + expf(x1 - x0));    // softmax component 0 (gumbel1)
    nm[n * 2 + 1] = 1.0f / (1.0f + expf(y0 - y1));    // softmax component 1 (gumbel2)
  }
}

__global__ void k_batchsum(const float* __restrict__ x, const int* __restrict__ batch,
                           float* __restrict__ gmean) {
  long t = blockIdx.x * (long)blockDim.x + threadIdx.x;
  const long total = N_ * 128;
  for (; t < total; t += (long)gridDim.x * blockDim.x) {
    const long n = t >> 7; const int h = (int)(t & 127);
    atomicAdd(&gmean[(long)batch[n] * 128 + h], x[t]);
  }
}

__global__ void k_gmean_fin(float* __restrict__ gmean, const float* __restrict__ cnt,
                            float* __restrict__ epsb) {
  const int b = blockIdx.x, h = threadIdx.x;
  __shared__ float red[128];
  float v = gmean[(long)b * 128 + h] / fmaxf(cnt[b], 1.0f);
  gmean[(long)b * 128 + h] = v;
  red[h] = v; __syncthreads();
  for (int s = 64; s > 0; s >>= 1) { if (h < s) red[h] += red[h + s]; __syncthreads(); }
  if (h == 0) epsb[b] = red[0] * (1.0f / 128.0f);
}

__global__ void k_mix(const float* __restrict__ x, const float* __restrict__ nm,
                      const float* __restrict__ epsb, const int* __restrict__ batch,
                      float* __restrict__ xs, float* __restrict__ xc) {
  long t = blockIdx.x * (long)blockDim.x + threadIdx.x;
  const long total = N_ * 128;
  for (; t < total; t += (long)gridDim.x * blockDim.x) {
    const long n = t >> 7;
    const float ep = epsb[batch[n]];
    const float xv = x[t];
    const float n0 = nm[n * 2], n1 = nm[n * 2 + 1];
    xs[t] = n0 * xv + (1.0f - n0) * ep;
    xc[t] = n1 * xv + (1.0f - n1) * ep;
  }
}

// ---------------------------------------------------------------------------
// host orchestration
// ---------------------------------------------------------------------------
extern "C" void kernel_launch(void* const* d_in, const int* in_sizes, int n_in,
                              void* d_out, int out_size, void* d_ws, size_t ws_size,
                              hipStream_t stream) {
  (void)in_sizes; (void)n_in; (void)out_size; (void)ws_size;
  auto F = [&](int i) { return (const float*)d_in[i]; };
  auto I = [&](int i) { return (const int*)d_in[i]; };

  // ---- param leaves (setup_inputs insertion order, depth-first) ----
  EmbP emb; for (int i = 0; i < 9; ++i) emb.p[i] = F(i);
  const float *Lg[3], *Lb[3], *LW[3], *Lc[3];
  for (int l = 0; l < 3; ++l) { Lg[l]=F(9+4*l); Lb[l]=F(10+4*l); LW[l]=F(11+4*l); Lc[l]=F(12+4*l); }
  const float *na_g1=F(21),*na_b1=F(22),*na_W1=F(23),*na_c1=F(24),
              *na_g2=F(25),*na_b2=F(26),*na_W2=F(27),*na_c2=F(28);
  const float *ea_g1=F(29),*ea_b1=F(30),*ea_W1=F(31),*ea_c1=F(32),
              *ea_g2=F(33),*ea_b2=F(34),*ea_W2=F(35),*ea_c2=F(36);
  const float *bnc_g=F(37),*bnc_b=F(38),*bno_g=F(39),*bno_b=F(40);
  const float *Wc=F(41),*cc=F(42),*Wo=F(43),*co=F(44);
  const float *ca_g1=F(45),*ca_b1=F(46),*ca_W1=F(47),*ca_c1=F(48),
              *ca_g2=F(49),*ca_b2=F(50),*ca_W2=F(51),*ca_c2=F(52);
  const float *ra_g1=F(53),*ra_b1=F(54),*ra_W1=F(55),*ra_c1=F(56),
              *ra_g2=F(57),*ra_b2=F(58),*ra_W2=F(59),*ra_c2=F(60);
  const float *gum1=F(61),*gum2=F(62);
  const int *xat=I(63),*ei=I(64),*hei=I(65),*batch=I(66),*c2n=I(67),*cmn=I(68),*cmc=I(69);

  char* ws = (char*)d_ws;
  float* X    = (float*)(ws + OFF_X);
  float* B1   = (float*)(ws + OFF_B1);   // msg(hi) / xs
  float* B2   = (float*)(ws + OFF_B2);   // hacc(hi) / xc
  float* B3   = (float*)(ws + OFF_B3);   // hidC / msg2
  __bf16* HID = (__bf16*)(ws + OFF_HID);
  char* sm = ws + OFF_SM;
  float* degHI = (float*)(sm + SM_DEGHI);
  float* dinvHI= (float*)(sm + SM_DINVHI);
  float* csum  = (float*)(sm + SM_CSUM);
  float* csq   = (float*)(sm + SM_CSQ);
  float* aff   = (float*)(sm + SM_AFF);
  __bf16* wbf  = (__bf16*)(sm + SM_WBF);
  float* rcnt  = (float*)(sm + SM_RCNT);
  float* ccnt  = (float*)(sm + SM_CCNT);
  float* degS  = (float*)(sm + SM_DEGS);
  float* dinvS = (float*)(sm + SM_DINVS);
  float* nm    = (float*)(sm + SM_NM);
  float* ea    = (float*)(sm + SM_EA);
  float* gmean = (float*)(sm + SM_GMEAN);
  float* cnt   = (float*)(sm + SM_CNT);
  float* epsb  = (float*)(sm + SM_EPSB);
  float* Sxs   = (float*)(sm + SM_SXS);
  float* Sxc   = (float*)(sm + SM_SXC);
  float* Ssum  = (float*)(sm + SM_SSUM);
  float* hidB  = (float*)(sm + SM_HIDB);
  float* na    = (float*)(sm + SM_NA);
  float* out   = (float*)d_out;

  const int TB = 256;
  const int GB = 4096;   // grid-stride blocks for big loops
  auto zero = [&](float* p, long n) { k_fill<<<GB, TB, 0, stream>>>(p, n, 0.0f); };
  auto stats128 = [&](const float* Xp, long rows, const float* g, const float* b) {
    k_fill<<<1, 256, 0, stream>>>(csum, 256, 0.0f);
    k_fill<<<1, 256, 0, stream>>>(csq, 256, 0.0f);
    k_colstats_f32<<<1024, 128, 0, stream>>>(Xp, rows, csum, csq);
    k_aff<<<1, 128, 0, stream>>>(csum, csq, g, b, 1.0f / (float)rows, 128, aff);
  };

  // ---- stage 0: embeddings + clique pooling ----
  k_embed<<<(unsigned)N_, 128, 0, stream>>>(emb, xat, X);
  zero(X + N_ * 128, C_ * 128);
  k_pool<<<GB, TB, 0, stream>>>(cmn, cmc, X, X + N_ * 128);

  // hi-graph degree (fixed across layers): deg = indeg(dst) + 1
  k_fill<<<GB, TB, 0, stream>>>(degHI, NC_, 1.0f);
  k_count<<<GB, TB, 0, stream>>>(hei + EHI_, EHI_, degHI);
  k_rsqrt<<<GB, TB, 0, stream>>>(degHI, dinvHI, NC_);

  // ---- stage 1: 3 residual GCN layers on hi-graph ----
  for (int l = 0; l < 3; ++l) {
    stats128(X, NC_, Lg[l], Lb[l]);
    k_pack_w<<<64, TB, 0, stream>>>(LW[l], wbf, 128, 128);
    k_gemm_aff_wmma<<<dim3((unsigned)(NC_ / 16), 2), 32, 0, stream>>>(
        X, nullptr, nullptr, 128, 128, aff, wbf, nullptr, B1 /*msg*/, nullptr, 0);
    zero(B2, NC_ * 128);
    k_scatter<<<GB, TB, 0, stream>>>(hei, hei + EHI_, nullptr, 0, dinvHI, B1, B2, EHI_);
    k_hi_fin<<<GB, TB, 0, stream>>>(X, B2, B1, dinvHI, Lc[l]);
  }

  // ---- motif attention MLP on x[N:] ----
  stats128(X + N_ * 128, C_, na_g1, na_b1);
  k_pack_w<<<64, TB, 0, stream>>>(na_W1, wbf, 128, 128);
  k_gemm_aff_wmma<<<dim3((unsigned)(C_ / 16), 2), 32, 0, stream>>>(
      X + N_ * 128, nullptr, nullptr, 128, 128, aff, wbf, na_c1, B3 /*hidC*/, nullptr, 1);
  stats128(B3, C_, na_g2, na_b2);
  k_head<<<GB, TB, 0, stream>>>(B3, aff, na_W2, na_c2, 128, C_, na);

  // gumbel-softmax mixing coefficients
  k_nm<<<GB, TB, 0, stream>>>(na, c2n, gum1, gum2, nm);

  // per-batch mean -> eps
  zero(gmean, B_ * 128); zero(cnt, B_);
  k_count<<<GB, TB, 0, stream>>>(batch, N_, cnt);
  k_batchsum<<<GB, TB, 0, stream>>>(X, batch, gmean);
  k_gmean_fin<<<(unsigned)B_, 128, 0, stream>>>(gmean, cnt, epsb);

  // xs / xc
  k_mix<<<GB, TB, 0, stream>>>(X, nm, epsb, batch, B1 /*xs*/, B2 /*xc*/);

  // ---- edge attention: bn1 stats via endpoint counts, gather-fused WMMA GEMM ----
  zero(rcnt, N_); zero(ccnt, N_);
  k_count<<<GB, TB, 0, stream>>>(ei, E_, rcnt);
  k_count<<<GB, TB, 0, stream>>>(ei + E_, E_, ccnt);
  k_fill<<<1, 256, 0, stream>>>(csum, 256, 0.0f);
  k_fill<<<1, 256, 0, stream>>>(csq, 256, 0.0f);
  k_wcolstats<<<1024, 256, 0, stream>>>(X, rcnt, ccnt, csum, csq);
  k_aff<<<1, 256, 0, stream>>>(csum, csq, ea_g1, ea_b1, 1.0f / (float)E_, 256, aff);
  k_pack_w<<<256, TB, 0, stream>>>(ea_W1, wbf, 256, 256);
  k_gemm_aff_wmma<<<dim3((unsigned)(E_ / 16), 4), 32, 0, stream>>>(
      X, ei, ei + E_, 256, 256, aff, wbf, ea_c1, nullptr, HID, 1);
  k_fill<<<1, 256, 0, stream>>>(csum, 256, 0.0f);
  k_fill<<<1, 256, 0, stream>>>(csq, 256, 0.0f);
  k_colstats_bf16<<<2048, 256, 0, stream>>>(HID, E_, csum, csq);
  k_aff<<<1, 256, 0, stream>>>(csum, csq, ea_g2, ea_b2, 1.0f / (float)E_, 256, aff);
  k_edge_head<<<GB, TB, 0, stream>>>(HID, aff, ea_W2, ea_c2, ea);

  // ---- stage 2 GCNs (edge-weighted), fused with batch segment-sum ----
  for (int side = 0; side < 2; ++side) {
    float* xin   = (side == 0) ? B1 : B2;       // xs / xc
    float* hacc2 = xin;                         // reuse after GEMM consumes xin
    const float* g  = (side == 0) ? bnc_g : bno_g;
    const float* b  = (side == 0) ? bnc_b : bno_b;
    const float* W  = (side == 0) ? Wc : Wo;
    const float* bi = (side == 0) ? cc : co;
    float* S        = (side == 0) ? Sxs : Sxc;
    const float* ew = ea + side;                // edge_att[:, side], stride 2

    stats128(xin, N_, g, b);
    k_pack_w<<<64, TB, 0, stream>>>(W, wbf, 128, 128);
    k_gemm_aff_wmma<<<dim3((unsigned)(N_ / 16), 2), 32, 0, stream>>>(
        xin, nullptr, nullptr, 128, 128, aff, wbf, nullptr, B3 /*msg2*/, nullptr, 0);
    k_fill<<<GB, TB, 0, stream>>>(degS, N_, 1.0f);
    k_deg_ew<<<GB, TB, 0, stream>>>(ei + E_, ew, 2, E_, degS);
    k_rsqrt<<<GB, TB, 0, stream>>>(degS, dinvS, N_);
    zero(hacc2, N_ * 128);
    k_scatter<<<GB, TB, 0, stream>>>(ei, ei + E_, ew, 2, dinvS, B3, hacc2, E_);
    zero(S, B_ * 128);
    k_s2_fin<<<GB, TB, 0, stream>>>(hacc2, B3, dinvS, bi, batch, S);
  }

  // ---- classifier heads ----
  k_addvec<<<GB, TB, 0, stream>>>(Sxs, Sxc, Ssum, B_ * 128);

  // causal(xc_sum) -> out[0]
  stats128(Sxc, B_, ca_g1, ca_b1);
  k_pack_w<<<64, TB, 0, stream>>>(ca_W1, wbf, 128, 128);
  k_gemm_aff_wmma<<<dim3((unsigned)(B_ / 16), 2), 32, 0, stream>>>(
      Sxc, nullptr, nullptr, 128, 128, aff, wbf, ca_c1, hidB, nullptr, 1);
  stats128(hidB, B_, ca_g2, ca_b2);
  k_head<<<GB, TB, 0, stream>>>(hidB, aff, ca_W2, ca_c2, 128, B_, out);

  // random(xs_sum + xc_sum) -> out[1]
  stats128(Ssum, B_, ra_g1, ra_b1);
  k_pack_w<<<64, TB, 0, stream>>>(ra_W1, wbf, 128, 128);
  k_gemm_aff_wmma<<<dim3((unsigned)(B_ / 16), 2), 32, 0, stream>>>(
      Ssum, nullptr, nullptr, 128, 128, aff, wbf, ra_c1, hidB, nullptr, 1);
  stats128(hidB, B_, ra_g2, ra_b2);
  k_head<<<GB, TB, 0, stream>>>(hidB, aff, ra_W2, ra_c2, 128, B_, out + B_ * 2);
}
